// EdgeMPNN_19868518711816
// MI455X (gfx1250) — matrix operand order repaired and checked by reading, not compiled, not run on hardware
//
#include <hip/hip_runtime.h>

// ---------------------------------------------------------------------------
// EdgeMPNN for MI455X (gfx1250, wave32, WMMA).
// One workgroup (256 threads = 8 wave32) per batch graph.
// f16 WMMA 16x16x32 for ALL four GEMM families (pre, emb, attention-message,
// update), f32 accumulate. m + emb + messages LDS-resident (~276 KB LDS of
// the 320 KB/WG the ISA allows); e0 is the only global intermediate.
// Branch-free fast tanh/sigmoid (no ocml branchy expansions).
// ---------------------------------------------------------------------------

typedef __attribute__((ext_vector_type(16))) _Float16 v16h;
typedef __attribute__((ext_vector_type(8)))  _Float16 v8h;
typedef __attribute__((ext_vector_type(8)))  float    v8f;

#define B_    1024
#define N_    24
#define FN_   32
#define FE_   8
#define E_    64
#define OUT_  128
#define NE_   (N_ * N_)        // 576 edges
#define KCAT_ (2 * FN_ + FE_)  // 72
#define KPRE_ 96               // 72 padded to 3 k-tiles of 32
#define KUPD_ 128
#define PASSES_ 3
#define NTILES_ (NE_ / 16)     // 36 row tiles of 16 edges
#define NWAVES_ 8

// ---- branch-free activations ----------------------------------------------
__device__ __forceinline__ float fast_tanh(float x) {
  const float t = __expf(-2.0f * fabsf(x));
  const float r = (1.0f - t) * __builtin_amdgcn_rcpf(1.0f + t);
  return copysignf(r, x);
}
__device__ __forceinline__ float fast_sigmoid(float x) {
  return __builtin_amdgcn_rcpf(1.0f + __expf(-x));
}

// ---- WMMA fragment loaders (wave32 layouts per CDNA5 ISA 7.12.2) ----------

// B fragment: 32x16 f16 (KxN). Lanes 0-15 hold K={0..7,16..23},
// lanes 16-31 hold K={8..15,24..31}; column = lane & 15.
__device__ __forceinline__ v16h load_bfrag(const _Float16* W, int ld, int k0,
                                           int n0, int lane) {
  const int col = n0 + (lane & 15);
  const int kb  = k0 + ((lane & 16) ? 8 : 0);
  v16h b;
#pragma unroll
  for (int t = 0; t < 16; ++t) b[t] = W[(kb + (t < 8 ? t : t + 8)) * ld + col];
  return b;
}

// A fragment: 16x32 f16 (MxK), row-major source. Lane L<16: row L, K chunks
// {k0..k0+7, k0+16..k0+23}; lane L>=16: row L-16, chunks {k0+8.., k0+24..}.
__device__ __forceinline__ v16h load_afrag_rm(const _Float16* A, int ld,
                                              int row0, int k0, int lane) {
  const int row = row0 + (lane & 15);
  const int kb  = k0 + ((lane & 16) ? 8 : 0);
  const _Float16* p = A + (size_t)row * ld + kb;
  v8h lo = *reinterpret_cast<const v8h*>(p);
  v8h hi = *reinterpret_cast<const v8h*>(p + 16);
  v16h a;
#pragma unroll
  for (int t = 0; t < 8; ++t) { a[t] = lo[t]; a[8 + t] = hi[t]; }
  return a;
}

__device__ __forceinline__ v8f wmma_f16(v16h a, v16h b, v8f c) {
  return __builtin_amdgcn_wmma_f32_16x16x32_f16(false, a, false, b, (short)0,
                                                c, false, false);
}

__global__ __launch_bounds__(256) void edgempnn_kernel(
    const float* __restrict__ nodes, const float* __restrict__ edges,
    const float* __restrict__ W_pre, const float* __restrict__ b_pre,
    const float* __restrict__ W_emb, const float* __restrict__ b_emb,
    const float* __restrict__ w_att, const float* __restrict__ b_att,
    const float* __restrict__ W_upd, const float* __restrict__ b_upd,
    const float* __restrict__ W_gate, const float* __restrict__ b_gate,
    const float* __restrict__ W_embr, const float* __restrict__ b_embr,
    const float* __restrict__ W_out, const float* __restrict__ b_out,
    float* __restrict__ out, _Float16* __restrict__ g_e0) {
  // ---- LDS (~276 KB total; ISA allows up to 320 KB per workgroup) ----
  __shared__ __align__(16) _Float16 sWpre[KPRE_ * E_];  // 12288 B (72..95 pad0)
  __shared__ __align__(16) _Float16 sWemb[E_ * E_];     //  8192 B
  __shared__ __align__(16) _Float16 sWupd[KUPD_ * E_];  // 16384 B
  __shared__ __align__(16) _Float16 sM[NE_ * E_];       // 73728 B  edge memory
  __shared__ __align__(16) _Float16 sEmb[NE_ * E_];     // 73728 B
  __shared__ __align__(16) _Float16 sMsgF[NE_ * E_];    // 73728 B
  __shared__ float sAttW[E_], sBpre[E_], sBemb[E_], sBupd[E_];  // 1024 B
  __shared__ float sEnergy[NE_];  // 2304 B
  __shared__ float sAf[NE_];      // 2304 B
  __shared__ float sExpE[NE_];    // 2304 B  af-masked exp(energy)
  __shared__ float sTotJ[N_];     //   96 B  per-j exp totals
  __shared__ __align__(16) char sPoolRaw[16384];  // aliased pool

  const int tid  = threadIdx.x;
  const int wave = tid >> 5;
  const int lane = tid & 31;
  const int batch = blockIdx.x;

  // warm L2 for readout-only weights (global_prefetch_b8)
  if (tid == 0) {
    __builtin_prefetch(W_gate, 0, 1);
    __builtin_prefetch(W_embr, 0, 1);
    __builtin_prefetch(W_out, 0, 1);
  }

  _Float16* e0B = g_e0 + (size_t)batch * NE_ * E_;

  // Phase-A view of the pool: nodes f16 (1536 B) + edges f16 (9216 B)
  _Float16* sNodesH = (_Float16*)sPoolRaw;
  _Float16* sEdges  = (_Float16*)(sPoolRaw + 2048);

  // ---- Stage weights (f32 -> f16) and per-graph data ----
  for (int idx = tid; idx < KPRE_ * E_; idx += 256) {
    int k = idx >> 6, c = idx & 63;
    sWpre[idx] = (_Float16)((k < KCAT_) ? W_pre[k * E_ + c] : 0.0f);
  }
  for (int idx = tid; idx < E_ * E_; idx += 256)
    sWemb[idx] = (_Float16)W_emb[idx];
  for (int idx = tid; idx < KUPD_ * E_; idx += 256)
    sWupd[idx] = (_Float16)W_upd[idx];
  if (tid < E_) {
    sAttW[tid] = w_att[tid];
    sBpre[tid] = b_pre[tid];
    sBemb[tid] = b_emb[tid];
    sBupd[tid] = b_upd[tid];
  }
  for (int idx = tid; idx < N_ * FN_; idx += 256)
    sNodesH[idx] = (_Float16)nodes[(size_t)batch * N_ * FN_ + idx];
  for (int idx = tid; idx < NE_ * FE_; idx += 256)
    sEdges[idx] = (_Float16)edges[(size_t)batch * NE_ * FE_ + idx];
  __syncthreads();
  for (int e = tid; e < NE_; e += 256) {
    float s = 0.f;
#pragma unroll
    for (int f = 0; f < FE_; ++f) s += (float)sEdges[e * FE_ + f];
    sAf[e] = (s > 0.f) ? 1.0f : 0.0f;
  }
  __syncthreads();

  // ---- Phase A: e0 = tanh(cat @ W_pre + b_pre) * af ; m = 0 ----
  for (int t = wave; t < NTILES_; t += NWAVES_) {
    const int row0 = t * 16;
    const int arow = row0 + (lane & 15);
    const int ai = arow / N_, aj = arow % N_;
    const int kb0 = (lane & 16) ? 8 : 0;
    const _Float16* nI = sNodesH + ai * FN_ + kb0;
    const _Float16* nJ = sNodesH + aj * FN_ + kb0;
    const _Float16* ep = sEdges + (ai * N_ + aj) * FE_;
    v16h a0, a1, a2;
#pragma unroll
    for (int u = 0; u < 8; ++u) {
      a0[u] = nI[u];  a0[8 + u] = nI[16 + u];
      a1[u] = nJ[u];  a1[8 + u] = nJ[16 + u];
      const _Float16 ev = ep[u];
      a2[u]     = (lane & 16) ? (_Float16)0.0f : ev;
      a2[8 + u] = (_Float16)0.0f;
    }
    v8f acc[4] = {};
#pragma unroll
    for (int nt = 0; nt < 4; ++nt) {
      acc[nt] = wmma_f16(a0, load_bfrag(sWpre, E_, 0,  nt * 16, lane), acc[nt]);
      acc[nt] = wmma_f16(a1, load_bfrag(sWpre, E_, 32, nt * 16, lane), acc[nt]);
      acc[nt] = wmma_f16(a2, load_bfrag(sWpre, E_, 64, nt * 16, lane), acc[nt]);
    }
#pragma unroll
    for (int r = 0; r < 8; ++r) {
      const int row = row0 + r + ((lane & 16) ? 8 : 0);
      const int col = lane & 15;
      const float af = sAf[row];
#pragma unroll
      for (int nt = 0; nt < 4; ++nt) {
        const int c = nt * 16 + col;
        float v = fast_tanh(acc[nt][r] + sBpre[c]) * af;
        e0B[row * E_ + c] = (_Float16)v;
        sM[row * E_ + c]  = (_Float16)0.0f;
      }
    }
  }
  __threadfence_block();  // e0 stores (global) visible before per-pass reads
  __syncthreads();

  // ---- Message passes (all per-pass state in LDS) ----
  for (int pass = 0; pass < PASSES_; ++pass) {
    // B1a: per-edge attention energies (vectorized LDS f16 loads)
    const float batt = b_att[0];
    for (int e = tid; e < NE_; e += 256) {
      const v8h* mr = reinterpret_cast<const v8h*>(sM + e * E_);
      float s = batt;
#pragma unroll
      for (int k8 = 0; k8 < 8; ++k8) {
        const v8h mv = mr[k8];
#pragma unroll
        for (int u = 0; u < 8; ++u) s += (float)mv[u] * sAttW[k8 * 8 + u];
      }
      sEnergy[e] = s;
    }
    // B1b: emb = tanh(m @ W_emb + b_emb) (WMMA) -> LDS
    for (int t = wave; t < NTILES_; t += NWAVES_) {
      const int row0 = t * 16;
      v8f acc[4] = {};
#pragma unroll
      for (int kt = 0; kt < 2; ++kt) {
        v16h a = load_afrag_rm(sM, E_, row0, kt * 32, lane);
#pragma unroll
        for (int nt = 0; nt < 4; ++nt) {
          v16h b = load_bfrag(sWemb, E_, kt * 32, nt * 16, lane);
          acc[nt] = wmma_f16(a, b, acc[nt]);
        }
      }
#pragma unroll
      for (int r = 0; r < 8; ++r) {
        const int row = row0 + r + ((lane & 16) ? 8 : 0);
        const int col = lane & 15;
#pragma unroll
        for (int nt = 0; nt < 4; ++nt) {
          const int c = nt * 16 + col;
          sEmb[row * E_ + c] = (_Float16)fast_tanh(acc[nt][r] + sBemb[c]);
        }
      }
    }
    __syncthreads();

    // B2a: af-masked exp(energy) per edge, then per-j totals.
    // softmax (shift-free; |energy| <= ||w_att||_1, safe in f32):
    //   att[i,j,k] = expw[j,k]*(k!=i) / (tot_j - expw[j,i])
    for (int e = tid; e < NE_; e += 256)
      sExpE[e] = (sAf[e] > 0.f) ? __expf(sEnergy[e]) : 0.f;
    __syncthreads();
    if (tid < N_) {
      float s = 0.f;
#pragma unroll
      for (int k = 0; k < N_; ++k) s += sExpE[tid * N_ + k];
      sTotJ[tid] = s;
    }
    __syncthreads();

    // B2b: message GEMMs via WMMA: for each j,
    //   message[i, j, :] = (A_j[32x32pad] @ emb_j[32x64pad]) * rowscale
    // A_j[i][k] = (k!=i) ? expw[j,k] : 0 ;  normalization folded into D store.
    for (int j = wave; j < N_; j += NWAVES_) {
      v16h aF[2];
#pragma unroll
      for (int mt = 0; mt < 2; ++mt) {
        const int i = mt * 16 + (lane & 15);
        const int kb = (lane & 16) ? 8 : 0;
        v16h a;
#pragma unroll
        for (int u = 0; u < 8; ++u) {
          const int k0 = kb + u;            // always < 24
          const int k1 = kb + 16 + u;       // >= 24 on lanes 16..31
          const int k1c = (lane & 16) ? 0 : k1;
          const float w0 = sExpE[j * N_ + k0];
          float w1 = sExpE[j * N_ + k1c];
          w1 = (lane & 16) ? 0.f : w1;
          a[u]     = (_Float16)((k0 == i) ? 0.f : w0);
          a[8 + u] = (_Float16)((k1 == i) ? 0.f : w1);
        }
        aF[mt] = a;
      }
      v8f acc[2][4] = {};
#pragma unroll
      for (int nt = 0; nt < 4; ++nt) {
        const int col = nt * 16 + (lane & 15);
        const int kb = (lane & 16) ? 8 : 0;
        v16h b;
#pragma unroll
        for (int t = 0; t < 16; ++t) {
          const int k = kb + (t < 8 ? t : t + 8);
          const int kc = (k < N_) ? k : 0;
          const _Float16 vv = sEmb[(j * N_ + kc) * E_ + col];
          b[t] = (k < N_) ? vv : (_Float16)0.0f;
        }
        acc[0][nt] = wmma_f16(aF[0], b, acc[0][nt]);
        acc[1][nt] = wmma_f16(aF[1], b, acc[1][nt]);
      }
      const float totj = sTotJ[j];
#pragma unroll
      for (int mt = 0; mt < 2; ++mt) {
        const bool rowok = (mt == 0) || ((lane & 16) == 0);
#pragma unroll
        for (int r = 0; r < 8; ++r) {
          if (rowok) {
            const int i = mt * 16 + r + ((lane & 16) ? 8 : 0);  // < 24 here
            const float denom = totj - sExpE[j * N_ + i];
            const float iv =
                (denom > 1e-20f) ? __builtin_amdgcn_rcpf(denom) : 0.f;
            const int col = lane & 15;
            const int e = i * N_ + j;
#pragma unroll
            for (int nt = 0; nt < 4; ++nt)
              sMsgF[e * E_ + nt * 16 + col] =
                  (_Float16)(acc[mt][nt][r] * iv);
          }
        }
      }
    }
    __syncthreads();

    // B2c: m = tanh([e0 | message] @ W_upd + b_upd) * af (WMMA, K=128)
    for (int t = wave; t < NTILES_; t += NWAVES_) {
      const int row0 = t * 16;
      v8f acc[4] = {};
#pragma unroll
      for (int kt = 0; kt < 2; ++kt) {
        v16h a = load_afrag_rm(e0B, E_, row0, kt * 32, lane);
#pragma unroll
        for (int nt = 0; nt < 4; ++nt) {
          v16h b = load_bfrag(sWupd, E_, kt * 32, nt * 16, lane);
          acc[nt] = wmma_f16(a, b, acc[nt]);
        }
      }
#pragma unroll
      for (int kt = 2; kt < 4; ++kt) {
        v16h a = load_afrag_rm(sMsgF, E_, row0, (kt - 2) * 32, lane);
#pragma unroll
        for (int nt = 0; nt < 4; ++nt) {
          v16h b = load_bfrag(sWupd, E_, kt * 32, nt * 16, lane);
          acc[nt] = wmma_f16(a, b, acc[nt]);
        }
      }
#pragma unroll
      for (int r = 0; r < 8; ++r) {
        const int row = row0 + r + ((lane & 16) ? 8 : 0);
        const int col = lane & 15;
        const float af = sAf[row];
#pragma unroll
        for (int nt = 0; nt < 4; ++nt) {
          const int c = nt * 16 + col;
          sM[row * E_ + c] = (_Float16)(fast_tanh(acc[nt][r] + sBupd[c]) * af);
        }
      }
    }
    __syncthreads();
  }

  // ---- Readout ----
  float* sGS      = (float*)sPoolRaw;           // [24][64]
  float* sContrib = (float*)(sPoolRaw + 6144);  // [24][64]
  for (int v = tid; v < N_ * 8; v += 256) {     // 192 items, v8h LDS loads
    const int i = v >> 3, c8 = v & 7;
    float s[8] = {};
    for (int j = 0; j < N_; ++j) {
      const v8h mv =
          *reinterpret_cast<const v8h*>(sM + (i * N_ + j) * E_ + c8 * 8);
#pragma unroll
      for (int u = 0; u < 8; ++u) s[u] += (float)mv[u];
    }
#pragma unroll
    for (int u = 0; u < 8; ++u) sGS[i * E_ + c8 * 8 + u] = s[u];
  }
  __syncthreads();
  for (int v = tid; v < N_ * E_; v += 256) {
    const int i = v >> 6, c = v & 63;
    float deg = 0.f;
#pragma unroll
    for (int j = 0; j < N_; ++j) deg += sAf[i * N_ + j];
    const float nm = (deg > 0.f) ? 1.0f : 0.0f;
    float gsum = b_gate[c], esum = b_embr[c];
#pragma unroll
    for (int k = 0; k < E_; ++k) {
      const float g = sGS[i * E_ + k];
      gsum += g * (W_gate[k * E_ + c] + W_gate[(E_ + k) * E_ + c]);
      esum += g * W_embr[k * E_ + c];
    }
    sContrib[v] = nm * fast_sigmoid(gsum) * fast_tanh(esum);
  }
  __syncthreads();
  float* sGraph = sEnergy;  // reuse (64 floats)
  if (tid < E_) {
    float s = 0.f;
#pragma unroll
    for (int i = 0; i < N_; ++i) s += sContrib[i * E_ + tid];
    sGraph[tid] = s;
  }
  __syncthreads();
  if (tid < OUT_) {
    float s = b_out[tid];
#pragma unroll
    for (int c = 0; c < E_; ++c) s += sGraph[c] * W_out[c * OUT_ + tid];
    out[(size_t)batch * OUT_ + tid] = s;
  }
}

extern "C" void kernel_launch(void* const* d_in, const int* in_sizes, int n_in,
                              void* d_out, int out_size, void* d_ws,
                              size_t ws_size, hipStream_t stream) {
  (void)in_sizes; (void)n_in; (void)out_size; (void)ws_size;
  const float* nodes  = (const float*)d_in[0];
  const float* edges  = (const float*)d_in[1];
  const float* W_pre  = (const float*)d_in[2];
  const float* b_pre  = (const float*)d_in[3];
  const float* W_emb  = (const float*)d_in[4];
  const float* b_emb  = (const float*)d_in[5];
  const float* w_att  = (const float*)d_in[6];
  const float* b_att  = (const float*)d_in[7];
  const float* W_upd  = (const float*)d_in[8];
  const float* b_upd  = (const float*)d_in[9];
  const float* W_gate = (const float*)d_in[10];
  const float* b_gate = (const float*)d_in[11];
  const float* W_embr = (const float*)d_in[12];
  const float* b_embr = (const float*)d_in[13];
  const float* W_out  = (const float*)d_in[14];
  const float* b_out  = (const float*)d_in[15];
  float* out = (float*)d_out;

  // Workspace: e0 in f16, [B, 576, 64] (~75 MB; L2-resident working set).
  _Float16* g_e0 = (_Float16*)d_ws;

  edgempnn_kernel<<<dim3(B_), dim3(256), 0, stream>>>(
      nodes, edges, W_pre, b_pre, W_emb, b_emb, w_att, b_att, W_upd, b_upd,
      W_gate, b_gate, W_embr, b_embr, W_out, b_out, out, g_e0);
}
